// LS_GNN_618475290910
// MI455X (gfx1250) — compile-verified, matrix operand order checked by reading, not run on hardware
//
#include <hip/hip_runtime.h>
#include <math.h>

#define BB 8
#define NNODE 10000
#define EE 320000
#define HIST 24
#define PRED 12
#define IN_DIM 17
#define GNN_OUT 13
#define HID 64

#define ROWS (BB * EE)        // 2,560,000 edge-batch rows
#define NTILES (ROWS / 16)    // 160,000 16-row tiles
#define EW_WAVES 8
#define EW_TPW 10             // tiles per wave; 160000/(8*10)=2000 blocks exactly

typedef __attribute__((ext_vector_type(16))) _Float16 v16h;
typedef __attribute__((ext_vector_type(8)))  float    v8f;

// fast sigmoid: v_exp_f32 + v_rcp_f32 (avoids IEEE div_scale/div_fmas chain)
__device__ __forceinline__ float sigmoidf(float x) {
    return __builtin_amdgcn_rcpf(1.0f + __expf(-x));
}
// fast tanh via sigmoid identity: tanh(x) = 2*sigmoid(2x) - 1
__device__ __forceinline__ float tanh_fast(float x) {
    return __fmaf_rn(2.0f, __builtin_amdgcn_rcpf(1.0f + __expf(-2.0f * x)), -1.0f);
}

// ---------------- edge_attr normalization (mean/std, ddof=1) ----------------
__global__ void ea_stats_kernel(const float* __restrict__ ea, float* __restrict__ stats) {
    float s = 0.f, s2 = 0.f;
    for (int i = blockIdx.x * blockDim.x + threadIdx.x; i < EE; i += gridDim.x * blockDim.x) {
        float v = ea[i]; s += v; s2 += v * v;
    }
    #pragma unroll
    for (int off = 16; off > 0; off >>= 1) {
        s  += __shfl_xor(s,  off, 32);
        s2 += __shfl_xor(s2, off, 32);
    }
    if ((threadIdx.x & 31) == 0) { atomicAdd(&stats[0], s); atomicAdd(&stats[1], s2); }
}

__global__ void ea_norm_kernel(const float* __restrict__ ea, const float* __restrict__ stats,
                               float* __restrict__ ean) {
    int i = blockIdx.x * blockDim.x + threadIdx.x;
    if (i >= EE) return;
    float mean = stats[0] / (float)EE;
    float var  = (stats[1] - stats[0] * mean) / (float)(EE - 1);
    ean[i] = (ea[i] - mean) * rsqrtf(var);
}

// ---------------- convert edge-MLP weights to padded f16 ----------------
// w1h: 64 x 32 (K-padded from 35x32), w2h: 32 x 32 (N-padded from 32x30)
__global__ void convert_w_kernel(const float* __restrict__ w1, const float* __restrict__ w2,
                                 _Float16* __restrict__ w1h, _Float16* __restrict__ w2h) {
    int i = blockIdx.x * blockDim.x + threadIdx.x;
    if (i < 64 * 32) {
        int k = i >> 5, n = i & 31;
        w1h[i] = (_Float16)((k < 35) ? w1[k * 32 + n] : 0.0f);
    }
    if (i < 32 * 32) {
        int k = i >> 5, n = i & 31;
        w2h[i] = (_Float16)((n < 30) ? w2[k * 30 + n] : 0.0f);
    }
}

// ---------------- init xn from last hist step ----------------
__global__ void init_xn_kernel(const float* __restrict__ t2m, float* __restrict__ xn) {
    int i = blockIdx.x * blockDim.x + threadIdx.x;
    if (i >= BB * NNODE) return;
    int b = i / NNODE, n = i - b * NNODE;
    xn[i] = t2m[((size_t)b * HIST + (HIST - 1)) * NNODE + n];
}

// ---------------- build x = concat([xn, feature_t]) : (B,N,17) ----------------
__global__ void build_x_kernel(const float* __restrict__ xn, const float* __restrict__ feat,
                               int t, float* __restrict__ x) {
    int i = blockIdx.x * blockDim.x + threadIdx.x;
    if (i >= BB * NNODE) return;
    int b = i / NNODE, n = i - b * NNODE;
    float* xr = x + (size_t)i * IN_DIM;
    xr[0] = xn[i];
    const float* f = feat + (((size_t)b * (HIST + PRED) + (HIST + t)) * NNODE + n) * (IN_DIM - 1);
    #pragma unroll
    for (int c = 0; c < IN_DIM - 1; ++c) xr[1 + c] = f[c];
}

// ---------------- edge MLP (WMMA) + scatter ----------------
__global__ __launch_bounds__(256) void edge_mlp_kernel(
    const float* __restrict__ x, const int* __restrict__ src, const int* __restrict__ tgt,
    const float* __restrict__ ean, const _Float16* __restrict__ w1h,
    const _Float16* __restrict__ w2h, const float* __restrict__ b1,
    const float* __restrict__ b2, float* __restrict__ agg)
{
    __shared__ _Float16 sA[EW_WAVES][16][64];   // layer-1 input tile (K padded to 64)
    __shared__ _Float16 sH[EW_WAVES][16][32];   // layer-1 output tile
    __shared__ int      sMeta[EW_WAVES][16][3]; // b, src, tgt per row

    const int lane  = threadIdx.x & 31;
    const int wv    = threadIdx.x >> 5;
    const int m     = lane & 15;
    const int kbase = (lane >= 16) ? 8 : 0;
    const int ncol  = lane & 15;

    // zero the whole A tile once; K cols >= 35 stay zero across all iterations
    for (int i = lane; i < 16 * 64; i += 32)
        (&sA[wv][0][0])[i] = (_Float16)0.0f;

    // preload B fragments (ISA B layout mirrors A: col = lane&15, K per element)
    v16h b1f[2][2];  // [ntile][kstep]
    v16h b2f[2];
    #pragma unroll
    for (int nt = 0; nt < 2; ++nt) {
        #pragma unroll
        for (int ks = 0; ks < 2; ++ks) {
            #pragma unroll
            for (int i = 0; i < 16; ++i) {
                int k = ks * 32 + kbase + (i < 8 ? i : i + 8);
                b1f[nt][ks][i] = w1h[k * 32 + nt * 16 + ncol];
            }
        }
        #pragma unroll
        for (int i = 0; i < 16; ++i) {
            int k = kbase + (i < 8 ? i : i + 8);
            b2f[nt][i] = w2h[k * 32 + nt * 16 + ncol];
        }
    }
    float bias1[2], bias2[2];
    bias1[0] = b1[ncol];
    bias1[1] = b1[16 + ncol];
    bias2[0] = b2[ncol];
    bias2[1] = (16 + ncol < 30) ? b2[16 + ncol] : 0.0f;

    const int wgid = blockIdx.x * EW_WAVES + wv;
    for (int it = 0; it < EW_TPW; ++it) {
        const int tile = wgid * EW_TPW + it;
        const long row0 = (long)tile * 16;

        // gather: 2 lanes per row (half 0 = src features, half 1 = tgt + ea)
        {
            const int r = lane >> 1;
            const int half = lane & 1;
            const long row = row0 + r;
            const int b = (int)(row / EE);
            const int e = (int)(row - (long)b * EE);
            const int sn = src[e], tn = tgt[e];
            if (half == 0) {
                const float* xs = x + ((size_t)b * NNODE + sn) * IN_DIM;
                #pragma unroll
                for (int c = 0; c < IN_DIM; ++c) sA[wv][r][c] = (_Float16)xs[c];
                sMeta[wv][r][0] = b; sMeta[wv][r][1] = sn; sMeta[wv][r][2] = tn;
            } else {
                const float* xt = x + ((size_t)b * NNODE + tn) * IN_DIM;
                #pragma unroll
                for (int c = 0; c < IN_DIM; ++c) sA[wv][r][IN_DIM + c] = (_Float16)xt[c];
                sA[wv][r][2 * IN_DIM] = (_Float16)ean[e];
            }
        }
        __syncthreads();

        // A fragments (16-bit A layout, §7.12.2)
        v16h a0, a1;
        #pragma unroll
        for (int i = 0; i < 16; ++i) {
            int k = kbase + (i < 8 ? i : i + 8);
            a0[i] = sA[wv][m][k];
            a1[i] = sA[wv][m][32 + k];
        }

        // layer 1: 16x64 @ 64x32
        v8f acc0 = {}, acc1 = {};
        acc0 = __builtin_amdgcn_wmma_f32_16x16x32_f16(false, a0, false, b1f[0][0], (short)0, acc0, false, false);
        acc0 = __builtin_amdgcn_wmma_f32_16x16x32_f16(false, a1, false, b1f[0][1], (short)0, acc0, false, false);
        acc1 = __builtin_amdgcn_wmma_f32_16x16x32_f16(false, a0, false, b1f[1][0], (short)0, acc1, false, false);
        acc1 = __builtin_amdgcn_wmma_f32_16x16x32_f16(false, a1, false, b1f[1][1], (short)0, acc1, false, false);

        // sigmoid + bias, spill D (N-across-lanes) to LDS to re-load as A (K-per-lane)
        {
            const int mb = (lane >= 16) ? 8 : 0;
            #pragma unroll
            for (int v = 0; v < 8; ++v) {
                sH[wv][mb + v][ncol]      = (_Float16)sigmoidf(acc0[v] + bias1[0]);
                sH[wv][mb + v][16 + ncol] = (_Float16)sigmoidf(acc1[v] + bias1[1]);
            }
        }
        __syncthreads();

        v16h a2;
        #pragma unroll
        for (int i = 0; i < 16; ++i) {
            int k = kbase + (i < 8 ? i : i + 8);
            a2[i] = sH[wv][m][k];
        }

        // layer 2: 16x32 @ 32x32 (cols >= 30 are zero-padded)
        v8f o0 = {}, o1 = {};
        o0 = __builtin_amdgcn_wmma_f32_16x16x32_f16(false, a2, false, b2f[0], (short)0, o0, false, false);
        o1 = __builtin_amdgcn_wmma_f32_16x16x32_f16(false, a2, false, b2f[1], (short)0, o1, false, false);

        // scatter: +h to tgt, -h to src
        {
            const int n1 = 16 + ncol;
            const int mb = (lane >= 16) ? 8 : 0;
            #pragma unroll
            for (int v = 0; v < 8; ++v) {
                const int r  = mb + v;
                const int b  = sMeta[wv][r][0];
                const int sn = sMeta[wv][r][1];
                const int tn = sMeta[wv][r][2];
                float* aggT = agg + ((size_t)b * NNODE + tn) * 30;
                float* aggS = agg + ((size_t)b * NNODE + sn) * 30;
                const float h0 = sigmoidf(o0[v] + bias2[0]);
                atomicAdd(&aggT[ncol],  h0);
                atomicAdd(&aggS[ncol], -h0);
                if (n1 < 30) {
                    const float h1 = sigmoidf(o1[v] + bias2[1]);
                    atomicAdd(&aggT[n1],  h1);
                    atomicAdd(&aggS[n1], -h1);
                }
            }
        }
        __syncthreads();
    }
}

// ---------------- node MLP: sigmoid(agg @ node_w + node_b) ----------------
__global__ void node_mlp_kernel(const float* __restrict__ agg, const float* __restrict__ nw,
                                const float* __restrict__ nb, float* __restrict__ g) {
    int i = blockIdx.x * blockDim.x + threadIdx.x;
    if (i >= BB * NNODE) return;
    const float* ar = agg + (size_t)i * 30;
    float a[30];
    #pragma unroll
    for (int k = 0; k < 30; ++k) a[k] = ar[k];
    float* gr = g + (size_t)i * GNN_OUT;
    #pragma unroll
    for (int o = 0; o < GNN_OUT; ++o) {
        float acc = nb[o];
        #pragma unroll
        for (int k = 0; k < 30; ++k) acc += a[k] * nw[k * GNN_OUT + o];
        gr[o] = sigmoidf(acc);
    }
}

// ---------------- GRU + output projection ----------------
__global__ __launch_bounds__(256) void gru_kernel(
    const float* __restrict__ g, const float* __restrict__ x,
    const float* __restrict__ wih, const float* __restrict__ whh,
    const float* __restrict__ bih, const float* __restrict__ bhh,
    const float* __restrict__ fcw, const float* __restrict__ fcb,
    float* __restrict__ h, float* __restrict__ xn, float* __restrict__ out, int t)
{
    __shared__ float s_wih[192 * 30];
    __shared__ float s_bih[192], s_bhh[192], s_fcw[64];
    for (int i = threadIdx.x; i < 192 * 30; i += blockDim.x) s_wih[i] = wih[i];
    for (int i = threadIdx.x; i < 192; i += blockDim.x) { s_bih[i] = bih[i]; s_bhh[i] = bhh[i]; }
    for (int i = threadIdx.x; i < 64; i += blockDim.x) s_fcw[i] = fcw[i];
    __syncthreads();

    int row = blockIdx.x * blockDim.x + threadIdx.x;
    if (row >= BB * NNODE) return;

    float xin[30];
    {
        const float* gr = g + (size_t)row * GNN_OUT;
        const float* xr = x + (size_t)row * IN_DIM;
        #pragma unroll
        for (int k = 0; k < GNN_OUT; ++k) xin[k] = gr[k];
        #pragma unroll
        for (int k = 0; k < IN_DIM; ++k) xin[GNN_OUT + k] = xr[k];
    }
    float hold[64];
    float* hr = h + (size_t)row * HID;
    #pragma unroll
    for (int k = 0; k < HID; ++k) hold[k] = hr[k];

    float xnv = fcb[0];
    for (int i = 0; i < HID; ++i) {
        float gir = s_bih[i], giz = s_bih[64 + i], gin = s_bih[128 + i];
        #pragma unroll
        for (int k = 0; k < 30; ++k) {
            gir += s_wih[i * 30 + k]          * xin[k];
            giz += s_wih[(64 + i) * 30 + k]   * xin[k];
            gin += s_wih[(128 + i) * 30 + k]  * xin[k];
        }
        float ghr = s_bhh[i], ghz = s_bhh[64 + i], ghn = s_bhh[128 + i];
        const float* wr = whh + (size_t)i * HID;
        const float* wz = whh + (size_t)(64 + i) * HID;
        const float* wn = whh + (size_t)(128 + i) * HID;
        #pragma unroll 8
        for (int k = 0; k < HID; ++k) {
            ghr += wr[k] * hold[k];
            ghz += wz[k] * hold[k];
            ghn += wn[k] * hold[k];
        }
        float rg = sigmoidf(gir + ghr);
        float zg = sigmoidf(giz + ghz);
        float ng = tanh_fast(gin + rg * ghn);
        float holdi = hr[i];              // old value (constant-index arrays stay in regs)
        float hv = (1.0f - zg) * ng + zg * holdi;
        hr[i] = hv;
        xnv += hv * s_fcw[i];
    }
    xn[row] = xnv;
    int b = row / NNODE, n = row - b * NNODE;
    out[((size_t)b * PRED + t) * NNODE + n] = xnv;
}

extern "C" void kernel_launch(void* const* d_in, const int* in_sizes, int n_in,
                              void* d_out, int out_size, void* d_ws, size_t ws_size,
                              hipStream_t stream) {
    const float* t2m  = (const float*)d_in[0];
    const float* feat = (const float*)d_in[1];
    const int*   ei   = (const int*)d_in[2];
    const float* ea   = (const float*)d_in[3];
    const float* w1   = (const float*)d_in[4];
    const float* b1   = (const float*)d_in[5];
    const float* w2   = (const float*)d_in[6];
    const float* b2   = (const float*)d_in[7];
    const float* nw   = (const float*)d_in[8];
    const float* nb   = (const float*)d_in[9];
    const float* wih  = (const float*)d_in[10];
    const float* whh  = (const float*)d_in[11];
    const float* bih  = (const float*)d_in[12];
    const float* bhh  = (const float*)d_in[13];
    const float* fcw  = (const float*)d_in[14];
    const float* fcb  = (const float*)d_in[15];
    float* out = (float*)d_out;
    const int* src = ei;
    const int* tgt = ei + EE;

    char* ws = (char*)d_ws;
    size_t off = 0;
    auto carve = [&](size_t bytes) -> void* {
        void* p = ws + off;
        off = (off + bytes + 255) & ~(size_t)255;
        return p;
    };
    float*    stats = (float*)carve(2 * sizeof(float));
    float*    ean   = (float*)carve((size_t)EE * sizeof(float));
    _Float16* w1h   = (_Float16*)carve(64 * 32 * sizeof(_Float16));
    _Float16* w2h   = (_Float16*)carve(32 * 32 * sizeof(_Float16));
    float*    xbuf  = (float*)carve((size_t)BB * NNODE * IN_DIM * sizeof(float));
    float*    aggb  = (float*)carve((size_t)BB * NNODE * 30 * sizeof(float));
    float*    gbuf  = (float*)carve((size_t)BB * NNODE * GNN_OUT * sizeof(float));
    float*    hbuf  = (float*)carve((size_t)BB * NNODE * HID * sizeof(float));
    float*    xnb   = (float*)carve((size_t)BB * NNODE * sizeof(float));

    const int rows = BB * NNODE;
    const int rblk = (rows + 255) / 256;

    (void)hipMemsetAsync(stats, 0, 2 * sizeof(float), stream);
    (void)hipMemsetAsync(hbuf, 0, (size_t)BB * NNODE * HID * sizeof(float), stream);
    ea_stats_kernel<<<512, 256, 0, stream>>>(ea, stats);
    ea_norm_kernel<<<(EE + 255) / 256, 256, 0, stream>>>(ea, stats, ean);
    convert_w_kernel<<<8, 256, 0, stream>>>(w1, w2, w1h, w2h);
    init_xn_kernel<<<rblk, 256, 0, stream>>>(t2m, xnb);

    for (int t = 0; t < PRED; ++t) {
        build_x_kernel<<<rblk, 256, 0, stream>>>(xnb, feat, t, xbuf);
        (void)hipMemsetAsync(aggb, 0, (size_t)BB * NNODE * 30 * sizeof(float), stream);
        edge_mlp_kernel<<<NTILES / (EW_WAVES * EW_TPW), 256, 0, stream>>>(
            xbuf, src, tgt, ean, w1h, w2h, b1, b2, aggb);
        node_mlp_kernel<<<rblk, 256, 0, stream>>>(aggb, nw, nb, gbuf);
        gru_kernel<<<rblk, 256, 0, stream>>>(gbuf, xbuf, wih, whh, bih, bhh,
                                             fcw, fcb, hbuf, xnb, out, t);
    }
}
